// QuantumQLSTM_65481071396696
// MI455X (gfx1250) — matrix operand order repaired and checked by reading, not compiled, not run on hardware
//
#include <hip/hip_runtime.h>
#include <hip/hip_bf16.h>
#include <hip/hip_fp16.h>

#define T_STEPS 512
#define BATCH   2048
#define DIN     128
#define NQ      8
#define MROWS   (T_STEPS * BATCH)   // 1,048,576 rows of the big GEMM
#define NTILES  (MROWS / 16)        // 65,536 16-row M-tiles

typedef __attribute__((ext_vector_type(16))) __bf16   v16bf;
typedef __attribute__((ext_vector_type(8)))  float    v8f;
typedef __attribute__((ext_vector_type(4)))  _Float16 v4h;

// ---------------------------------------------------------------------------
// Kernel 1: time-parallel input GEMM with bf16 WMMA.
//   Z[((t*B + b)*8 + q)*4 + g] = fp16( sum_k X[t,b,k] * W_g[k,q] ),  k<128
// Each wave: one 16-row M-tile, two 16x16 N-tiles, 4 x K=32 bf16 WMMA steps.
// Weight fragments (128x32 bf16) are wave-invariant: built once, kept in VGPRs.
// ---------------------------------------------------------------------------
__global__ void qlstm_gemm_wmma(const float* __restrict__ X,
                                const float* __restrict__ Wf,
                                const float* __restrict__ Wi,
                                const float* __restrict__ Wg,
                                const float* __restrict__ Wo,
                                _Float16* __restrict__ Z)
{
    const int lane  = threadIdx.x & 31;
    const int gwave = (int)((blockIdx.x * blockDim.x + threadIdx.x) >> 5);
    const int nwave = (int)((gridDim.x * blockDim.x) >> 5);

    const float* Wt[4] = {Wf, Wi, Wg, Wo};

    const int nl = lane & 15;   // N column (lo tile) / M row within tile
    const int lh = lane >> 4;   // lane-half selects K sub-block

    // B fragments, this code's contiguous k order — lanes 0-15 hold N=lane K=0..15,
    // lanes 16-31 hold N=lane-16 K=16..31 (2 bf16 per VGPR).
    v16bf bfrag[8];             // [kstep*2 + {lo,hi}]
#pragma unroll
    for (int s = 0; s < 4; ++s) {
        const int kbase = s * 32 + lh * 16;
#pragma unroll
        for (int half = 0; half < 2; ++half) {
            const int n = half * 16 + nl;
            const float* __restrict__ W = Wt[n >> 3];
            const int q = n & 7;
            v16bf v;
#pragma unroll
            for (int j = 0; j < 16; ++j)
                v[j] = (__bf16)W[(kbase + j) * NQ + q];
            bfrag[s * 2 + half] = v;
        }
    }

    // Z column offset inside a row: n=(g,q) stored at q*4+g (gate innermost).
    const int n0 = nl, n1 = 16 + nl;
    const int zoff0 = (n0 & 7) * 4 + (n0 >> 3);
    const int zoff1 = (n1 & 7) * 4 + (n1 >> 3);

    for (int tile = gwave; tile < NTILES; tile += nwave) {
        const float* __restrict__ xrow = X + (size_t)(tile * 16 + nl) * DIN;
        if (tile + nwave < NTILES)
            __builtin_prefetch(X + (size_t)((tile + nwave) * 16 + nl) * DIN, 0, 1);

        v8f c0 = {};
        v8f c1 = {};
#pragma unroll
        for (int s = 0; s < 4; ++s) {
            // A fragment, contiguous k order: elems 0..15 -> K = 32s + lh*16 + j (matches B).
            const int k0 = s * 32 + lh * 16;
            const float4 f0 = *(const float4*)(xrow + k0);
            const float4 f1 = *(const float4*)(xrow + k0 + 4);
            const float4 f2 = *(const float4*)(xrow + k0 + 8);
            const float4 f3 = *(const float4*)(xrow + k0 + 12);
            v16bf a;
            a[0]  = (__bf16)f0.x; a[1]  = (__bf16)f0.y; a[2]  = (__bf16)f0.z; a[3]  = (__bf16)f0.w;
            a[4]  = (__bf16)f1.x; a[5]  = (__bf16)f1.y; a[6]  = (__bf16)f1.z; a[7]  = (__bf16)f1.w;
            a[8]  = (__bf16)f2.x; a[9]  = (__bf16)f2.y; a[10] = (__bf16)f2.z; a[11] = (__bf16)f2.w;
            a[12] = (__bf16)f3.x; a[13] = (__bf16)f3.y; a[14] = (__bf16)f3.z; a[15] = (__bf16)f3.w;

            c0 = __builtin_amdgcn_wmma_f32_16x16x32_bf16(
                     false, a, false, bfrag[s * 2 + 0], (short)0, c0, false, false);
            c1 = __builtin_amdgcn_wmma_f32_16x16x32_bf16(
                     false, a, false, bfrag[s * 2 + 1], (short)0, c1, false, false);
        }

        // C layout: VGPR r, lanes 0-15 -> M=r, lanes 16-31 -> M=8+r; N=lane%16.
#pragma unroll
        for (int r = 0; r < 8; ++r) {
            const size_t mrow = (size_t)(tile * 16 + lh * 8 + r) * 32;
            Z[mrow + zoff0] = (_Float16)c0[r];
            Z[mrow + zoff1] = (_Float16)c1[r];
        }
    }
}

// ---------------------------------------------------------------------------
// Kernel 2: sequential LSTM recurrence, one thread per (batch, q) pair.
// 16,384 threads; per step each thread does the 4 gate pre-activations for its
// q, gathering the 8 h-components of its batch element via width-8 shuffles.
// Z row for (t,b,q) is 4 contiguous fp16 -> one 8-byte coalesced load,
// software-pipelined one step ahead.
// ---------------------------------------------------------------------------
__global__ void qlstm_recurrence(const _Float16* __restrict__ Z,
                                 const float* __restrict__ Wf, const float* __restrict__ bf_, const float* __restrict__ tf_,
                                 const float* __restrict__ Wi, const float* __restrict__ bi_, const float* __restrict__ ti_,
                                 const float* __restrict__ Wg, const float* __restrict__ bg_, const float* __restrict__ tg_,
                                 const float* __restrict__ Wo, const float* __restrict__ bo_, const float* __restrict__ to_,
                                 float* __restrict__ out)
{
    const int tid = blockIdx.x * blockDim.x + threadIdx.x;   // 0 .. B*NQ-1
    const int q   = tid & 7;

    // Recurrent weight columns for this q (rows 128..135 of each gate matrix).
    float whf[8], whi[8], whg[8], who[8];
#pragma unroll
    for (int j = 0; j < 8; ++j) {
        whf[j] = Wf[(DIN + j) * NQ + q];
        whi[j] = Wi[(DIN + j) * NQ + q];
        whg[j] = Wg[(DIN + j) * NQ + q];
        who[j] = Wo[(DIN + j) * NQ + q];
    }
    const float betaf = bf_[q] + tf_[q];
    const float betai = bi_[q] + ti_[q];
    const float betag = bg_[q] + tg_[q];
    const float betao = bo_[q] + to_[q];

    float h = 0.0f, c = 0.0f;

    v4h znext = *(const v4h*)(Z + (size_t)tid * 4);
    for (int t = 0; t < T_STEPS; ++t) {
        const v4h z = znext;
        if (t + 1 < T_STEPS)
            znext = *(const v4h*)(Z + (size_t)(t + 1) * (BATCH * 32) + (size_t)tid * 4);

        float af = (float)z[0] + betaf;
        float ai = (float)z[1] + betai;
        float ag = (float)z[2] + betag;
        float ao = (float)z[3] + betao;
#pragma unroll
        for (int j = 0; j < 8; ++j) {
            const float hj = __shfl(h, j, 8);   // h[b][j] from sibling lane
            af += hj * whf[j];
            ai += hj * whi[j];
            ag += hj * whg[j];
            ao += hj * who[j];
        }

        const float fg = 1.0f / (1.0f + __expf(-__cosf(af)));
        const float ig = 1.0f / (1.0f + __expf(-__cosf(ai)));
        const float gg = tanhf(__cosf(ag));
        const float og = 1.0f / (1.0f + __expf(-__cosf(ao)));

        c = fg * c + ig * gg;
        h = og * tanhf(c);

        out[(size_t)t * (BATCH * NQ) + tid] = h;   // outputs[t,b,q]
    }

    // Final (hx, cx) appended after outputs.
    out[(size_t)T_STEPS * BATCH * NQ + tid] = h;
    out[(size_t)T_STEPS * BATCH * NQ + BATCH * NQ + tid] = c;
}

// ---------------------------------------------------------------------------
extern "C" void kernel_launch(void* const* d_in, const int* in_sizes, int n_in,
                              void* d_out, int out_size, void* d_ws, size_t ws_size,
                              hipStream_t stream)
{
    const float* X   = (const float*)d_in[0];
    const float* Wf  = (const float*)d_in[1];
    const float* bf_ = (const float*)d_in[2];
    const float* tf_ = (const float*)d_in[3];
    const float* Wi  = (const float*)d_in[4];
    const float* bi_ = (const float*)d_in[5];
    const float* ti_ = (const float*)d_in[6];
    const float* Wg  = (const float*)d_in[7];
    const float* bg_ = (const float*)d_in[8];
    const float* tg_ = (const float*)d_in[9];
    const float* Wo  = (const float*)d_in[10];
    const float* bo_ = (const float*)d_in[11];
    const float* to_ = (const float*)d_in[12];

    _Float16* Z = (_Float16*)d_ws;   // T*B*32 fp16 = 64 MB staging
    float* out  = (float*)d_out;

    // 8192 blocks x 256 threads = 65,536 waves -> one 16-row M-tile per wave.
    qlstm_gemm_wmma<<<8192, 256, 0, stream>>>(X, Wf, Wi, Wg, Wo, Z);

    // 16,384 threads = 512 waves; recurrence independent per batch element.
    qlstm_recurrence<<<(BATCH * NQ) / 256, 256, 0, stream>>>(
        Z, Wf, bf_, tf_, Wi, bi_, ti_, Wg, bg_, tg_, Wo, bo_, to_, out);
}
